// PixelDSNTDistanceDoublePredict_20083267076471
// MI455X (gfx1250) — matrix operand order repaired and checked by reading, not compiled, not run on hardware
//
#include <hip/hip_runtime.h>
#include <hip/hip_bf16.h>
#include <math.h>

typedef __attribute__((ext_vector_type(16))) _Float16 v16h;
typedef __attribute__((ext_vector_type(8)))  float    v8f;

#define Bsz   32
#define Csz   2
#define Hsz   512
#define Wsz   512
#define HWsz  (Hsz * Wsz)      // 262144
#define NPAIR (Bsz * Csz)      // 64
#define S1    32               // pass-1 splits per pair
#define CHUNK1 (HWsz / S1)     // 8192 elements per pass-1 block
#define S3    8                // pass-3 blocks per pair (8 waves * 8 = 64 units = 32 strips x 2 col-halves)

// ---------------------------------------------------------------------------
// Pass 1: per (pair, split) chunk -> max(input), argmax(target) (first-index tiebreak)
// ---------------------------------------------------------------------------
__global__ __launch_bounds__(256) void k_pass1(const float* __restrict__ in,
                                               const float* __restrict__ tg,
                                               float* __restrict__ inMaxP,
                                               float* __restrict__ tValP,
                                               int*   __restrict__ tIdxP) {
    int pair  = blockIdx.x >> 5;   // /32
    int split = blockIdx.x & 31;
    const float* ip = in + (size_t)pair * HWsz + (size_t)split * CHUNK1;
    const float* tp = tg + (size_t)pair * HWsz + (size_t)split * CHUNK1;
    int t = threadIdx.x;

    float bmax = -INFINITY;
    float tbv  = -INFINITY;
    int   tbi  = 0x7fffffff;

    #pragma unroll
    for (int i = 0; i < CHUNK1 / 1024; ++i) {          // 8 iterations
        int e = (i * 256 + t) * 4;
        // prefetch next iteration's lines (global_prefetch_b8, no counter cost)
        if (i + 1 < CHUNK1 / 1024) {
            __builtin_prefetch(ip + e + 1024, 0, 1);
            __builtin_prefetch(tp + e + 1024, 0, 1);
        }
        float4 a = *(const float4*)(ip + e);
        float4 b = *(const float4*)(tp + e);
        bmax = fmaxf(bmax, fmaxf(fmaxf(a.x, a.y), fmaxf(a.z, a.w)));
        int gbase = split * CHUNK1 + e;
        float tv[4] = {b.x, b.y, b.z, b.w};
        #pragma unroll
        for (int j = 0; j < 4; ++j) {
            if (tv[j] > tbv || (tv[j] == tbv && (gbase + j) < tbi)) {
                tbv = tv[j]; tbi = gbase + j;
            }
        }
    }

    __shared__ float sm[256];
    __shared__ float sv[256];
    __shared__ int   si[256];
    sm[t] = bmax; sv[t] = tbv; si[t] = tbi;
    __syncthreads();
    for (int s = 128; s > 0; s >>= 1) {
        if (t < s) {
            sm[t] = fmaxf(sm[t], sm[t + s]);
            float ov = sv[t + s]; int oi = si[t + s];
            if (ov > sv[t] || (ov == sv[t] && oi < si[t])) { sv[t] = ov; si[t] = oi; }
        }
        __syncthreads();
    }
    if (t == 0) {
        inMaxP[blockIdx.x] = sm[0];
        tValP[blockIdx.x]  = sv[0];
        tIdxP[blockIdx.x]  = si[0];
    }
}

// ---------------------------------------------------------------------------
// Pass 2: merge the S1 partials of each pair
// ---------------------------------------------------------------------------
__global__ void k_pass2(const float* __restrict__ inMaxP,
                        const float* __restrict__ tValP,
                        const int*   __restrict__ tIdxP,
                        float* __restrict__ M, int* __restrict__ TI) {
    int p = threadIdx.x;
    if (p >= NPAIR) return;
    float m = -INFINITY, tv = -INFINITY;
    int ti = 0x7fffffff;
    for (int s = 0; s < S1; ++s) {
        m = fmaxf(m, inMaxP[p * S1 + s]);
        float v = tValP[p * S1 + s];
        int   i = tIdxP[p * S1 + s];
        if (v > tv || (v == tv && i < ti)) { tv = v; ti = i; }
    }
    M[p] = m; TI[p] = ti;
}

// ---------------------------------------------------------------------------
// Pass 3: streaming exp + WMMA-weighted reductions.
// Each wave owns one column-half (8 tiles of 16x32) of a 16-row strip.
// A = exp tile (16x32 f16); B fixed: col0 = relative K index (0..31), col1 = 1.
// C (f32) accumulates across the wave's tiles; the (cb+1)*rowsum term stays in VALU.
// ---------------------------------------------------------------------------
__global__ __launch_bounds__(256) void k_pass3(const float* __restrict__ in,
                                               const float* __restrict__ M,
                                               float* __restrict__ part) {
    int pair  = blockIdx.x / S3;
    int split = blockIdx.x % S3;
    const float* bp = in + (size_t)pair * HWsz;
    float mM = M[pair];

    int t    = threadIdx.x;
    int lane = t & 31;
    int wave = t >> 5;
    int m    = lane & 15;   // A-operand row / B-operand column for this lane
    int half = lane >> 4;
    int unit    = split * 8 + wave;   // 0..63
    int strip   = unit >> 1;          // 0..31 (16-row strip)
    int colhalf = unit & 1;           // which 8 column tiles
    int rowbase = strip * 16;

    // B operand (loop invariant): B[k, 0] = k,  B[k, 1] = 1, else 0.
    // Lane layout (16-bit B, 32x16): element i<8 holds K=8*half+i, i>=8 holds K=16+8*half+(i-8).
    v16h bmat;
    #pragma unroll
    for (int i = 0; i < 16; ++i) {
        float bv = 0.0f;
        if (m == 0)      bv = (float)((i < 8) ? (8 * half + i) : (16 + 8 * half + (i - 8)));
        else if (m == 1) bv = 1.0f;
        bmat[i] = (_Float16)bv;
    }

    const float* rowp = bp + (size_t)(rowbase + m) * Wsz;
    v8f   acc    = {};
    float sxbase = 0.0f;

    for (int tile = 0; tile < 8; ++tile) {
        int cb = (colhalf * 8 + tile) * 32;
        int c0 = cb + 8 * half;
        if (tile + 1 < 8) {                       // prefetch next tile
            __builtin_prefetch(rowp + c0 + 32, 0, 1);
        }
        float4 f0 = *(const float4*)(rowp + c0);
        float4 f1 = *(const float4*)(rowp + c0 + 4);
        float4 f2 = *(const float4*)(rowp + c0 + 16);
        float4 f3 = *(const float4*)(rowp + c0 + 20);

        float ex[16];
        ex[0]  = __expf(f0.x - mM); ex[1]  = __expf(f0.y - mM);
        ex[2]  = __expf(f0.z - mM); ex[3]  = __expf(f0.w - mM);
        ex[4]  = __expf(f1.x - mM); ex[5]  = __expf(f1.y - mM);
        ex[6]  = __expf(f1.z - mM); ex[7]  = __expf(f1.w - mM);
        ex[8]  = __expf(f2.x - mM); ex[9]  = __expf(f2.y - mM);
        ex[10] = __expf(f2.z - mM); ex[11] = __expf(f2.w - mM);
        ex[12] = __expf(f3.x - mM); ex[13] = __expf(f3.y - mM);
        ex[14] = __expf(f3.z - mM); ex[15] = __expf(f3.w - mM);

        float ts = 0.0f;
        v16h a;
        #pragma unroll
        for (int i = 0; i < 16; ++i) { ts += ex[i]; a[i] = (_Float16)ex[i]; }
        sxbase += (float)(cb + 1) * ts;   // absolute-column base contribution

        acc = __builtin_amdgcn_wmma_f32_16x16x32_f16(
                  false, a, false, bmat, (short)0, acc, false, false);
    }

    // Epilogue.  C/D layout: VGPR r -> row (r + 8*half) for this lane's half,
    // column = lane % 16.  Column 0 = relative-x-weighted row sums, column 1 = row sums.
    float sumc = 0.0f, ysum = 0.0f;
    int r0 = rowbase + 8 * half;
    #pragma unroll
    for (int r = 0; r < 8; ++r) {
        float cv = acc[r];
        sumc += cv;
        ysum += (float)(r0 + r + 1) * cv;
    }
    float v_sx = sxbase + ((m == 0) ? sumc : 0.0f);
    float v_s  = (m == 1) ? sumc : 0.0f;
    float v_sy = (m == 1) ? ysum : 0.0f;

    __shared__ float rs[256];
    __shared__ float rx[256];
    __shared__ float ry[256];
    rs[t] = v_s; rx[t] = v_sx; ry[t] = v_sy;
    __syncthreads();
    for (int s = 128; s > 0; s >>= 1) {
        if (t < s) { rs[t] += rs[t + s]; rx[t] += rx[t + s]; ry[t] += ry[t + s]; }
        __syncthreads();
    }
    if (t == 0) {
        part[blockIdx.x * 3 + 0] = rs[0];
        part[blockIdx.x * 3 + 1] = rx[0];
        part[blockIdx.x * 3 + 2] = ry[0];
    }
}

// ---------------------------------------------------------------------------
// Pass 4: finalize everything, write the 260 outputs.
// ---------------------------------------------------------------------------
__global__ void k_final(const float* __restrict__ part,
                        const int*   __restrict__ TI,
                        float* __restrict__ out) {
    __shared__ float spx[NPAIR], spy[NPAIR], stx[NPAIR], sty[NPAIR], sed[NPAIR];
    int p = threadIdx.x;
    if (p < NPAIR) {
        float S = 0.0f, SX = 0.0f, SY = 0.0f;
        for (int s = 0; s < S3; ++s) {
            int bi = p * S3 + s;
            S  += part[bi * 3 + 0];
            SX += part[bi * 3 + 1];
            SY += part[bi * 3 + 2];
        }
        float px = SX / S, py = SY / S;          // pixel coords (softmax * W/H folded in)
        int ti = TI[p];
        float tx = (float)(ti % Wsz + 1);
        float ty = (float)(ti / Wsz + 1);
        float dx = tx - px, dy = ty - py;
        float ed = sqrtf(dx * dx + dy * dy);
        out[4 + p * 2 + 0]       = px - 1.0f;    // pred_coords
        out[4 + p * 2 + 1]       = py - 1.0f;
        out[4 + 128 + p * 2 + 0] = tx - 1.0f;    // true_coords
        out[4 + 128 + p * 2 + 1] = ty - 1.0f;
        spx[p] = px; spy[p] = py; stx[p] = tx; sty[p] = ty; sed[p] = ed;
    }
    __syncthreads();
    if (p == 0) {
        float si = 0.0f, ss = 0.0f, sd = 0.0f;
        for (int b = 0; b < Bsz; ++b) {
            si += sed[2 * b];
            ss += sed[2 * b + 1];
            float vpx = spx[2 * b] - spx[2 * b + 1];
            float vpy = spy[2 * b] - spy[2 * b + 1];
            float vtx = stx[2 * b] - stx[2 * b + 1];
            float vty = sty[2 * b] - sty[2 * b + 1];
            float pd = sqrtf(vpx * vpx + vpy * vpy);
            float td = sqrtf(vtx * vtx + vty * vty);
            sd += fabsf(pd - td);
        }
        float bn = (float)Bsz;
        out[0] = si / bn;
        out[1] = ss / bn;
        out[2] = (si + ss) / bn;
        out[3] = sd / bn;
    }
}

// ---------------------------------------------------------------------------
extern "C" void kernel_launch(void* const* d_in, const int* in_sizes, int n_in,
                              void* d_out, int out_size, void* d_ws, size_t ws_size,
                              hipStream_t stream) {
    const float* in = (const float*)d_in[0];
    const float* tg = (const float*)d_in[1];
    float* out = (float*)d_out;

    char* ws = (char*)d_ws;
    float* inMaxP = (float*)(ws + 0);        // 2048 floats
    float* tValP  = (float*)(ws + 8192);     // 2048 floats
    int*   tIdxP  = (int*)  (ws + 16384);    // 2048 ints
    float* M      = (float*)(ws + 24576);    // 64 floats
    int*   TI     = (int*)  (ws + 24832);    // 64 ints
    float* part   = (float*)(ws + 25088);    // 512*3 floats

    k_pass1<<<NPAIR * S1, 256, 0, stream>>>(in, tg, inMaxP, tValP, tIdxP);
    k_pass2<<<1, 64, 0, stream>>>(inMaxP, tValP, tIdxP, M, TI);
    k_pass3<<<NPAIR * S3, 256, 0, stream>>>(in, M, part);
    k_final<<<1, 64, 0, stream>>>(part, TI, out);
}
